// EncoderLayer_83760452206932
// MI455X (gfx1250) — matrix operand order repaired and checked by reading, not compiled, not run on hardware
//
#include <hip/hip_runtime.h>
#include <hip/hip_bf16.h>

#define N_TOK  16384
#define DMODEL 512
#define NHEAD  8
#define DKH    64
#define DFF    2048
#define NTOP   384
#define NRAND  128
#define KTOT   512

typedef __bf16 bf16;
typedef __attribute__((ext_vector_type(16))) __bf16 bf16x16;
typedef __attribute__((ext_vector_type(8)))  __bf16 bf16x8;
typedef __attribute__((ext_vector_type(8)))  float  f32x8;
typedef __attribute__((ext_vector_type(4)))  unsigned u32x4;
typedef __attribute__((ext_vector_type(8)))  int    i32x8;
typedef __attribute__((ext_vector_type(4)))  int    i32x4;

__device__ __forceinline__ bf16 f2bf(float f) {
  unsigned u = __builtin_bit_cast(unsigned, f);
  unsigned short h = (unsigned short)((u + 0x7FFFu + ((u >> 16) & 1u)) >> 16);
  return __builtin_bit_cast(bf16, h);
}

// ---- WMMA fragment loaders (CDNA5 16-bit layouts, wave32) ------------------
// A 16x32 (MxK): lanes 0-15 -> M=lane, K {k0..k0+7} and {k0+16..k0+23};
//                lanes 16-31 -> M=lane-16, K {k0+8..k0+15} and {k0+24..k0+31}
__device__ __forceinline__ bf16x16 load_a_frag(const bf16* __restrict__ A, long lda,
                                               int row0, int k0, int lane) {
  int m    = lane & 15;
  int half = (lane >> 4) & 1;
  const bf16* p = A + (long)(row0 + m) * lda + k0 + half * 8;
  bf16x8 lo = *(const bf16x8*)(p);
  bf16x8 hi = *(const bf16x8*)(p + 16);
  return __builtin_shufflevector(lo, hi, 0,1,2,3,4,5,6,7,8,9,10,11,12,13,14,15);
}

// B 32x16 (KxN) from Bt (N-major, Bt[n][k]): lanes 0-15 -> K {k0..k0+7} and {k0+16..k0+23} at col n,
// lanes 16-31 -> K {k0+8..k0+15} and {k0+24..k0+31}.
__device__ __forceinline__ bf16x16 load_b_frag(const bf16* __restrict__ Bt, long ldbt,
                                               int col0, int k0, int lane) {
  int n    = lane & 15;
  int half = (lane >> 4) & 1;
  const bf16* p = Bt + (long)(col0 + n) * ldbt + k0 + half * 8;
  bf16x8 lo = *(const bf16x8*)(p);
  bf16x8 hi = *(const bf16x8*)(p + 16);
  return __builtin_shufflevector(lo, hi, 0,1,2,3,4,5,6,7,8,9,10,11,12,13,14,15);
}

// ---- Tensor Data Mover: 2-D bf16 tile (global, row-major) -> LDS -----------
// D# per CDNA5 ISA ch.8: group0 = {count/type, lds_addr, global_addr},
// group1 = {data_size, tensor dims, tile dims, dim0 stride}. 2-D tile:
// tile_dim2 = 0, groups 2/3 zero.
__device__ __forceinline__ void tdm_load_2d(unsigned lds_addr, const bf16* gptr,
                                            unsigned tileK, unsigned tileRows,
                                            unsigned long long strideElems) {
  unsigned long long ga = (unsigned long long)(uintptr_t)gptr;
  u32x4 g0;
  g0[0] = 1u;                                                  // count=1, user D#
  g0[1] = lds_addr;                                            // lds_addr[31:0]
  g0[2] = (unsigned)ga;                                        // global_addr lo
  g0[3] = (unsigned)((ga >> 32) & 0x01FFFFFFull) | (2u << 30); // addr hi | type=2
  i32x8 g1;
  unsigned long long s0 = strideElems;                         // dim0 stride (elems)
  g1[0] = (int)(1u << 16);                                     // data_size=1 (2 bytes)
  g1[1] = (int)((tileK & 0xFFFFu) << 16);                      // tensor_dim0[15:0]
  g1[2] = (int)((tileK >> 16) | ((tileRows & 0xFFFFu) << 16)); // td0 hi | tensor_dim1 lo
  g1[3] = (int)((tileRows >> 16) | (tileK << 16));             // td1 hi | tile_dim0
  g1[4] = (int)(tileRows & 0xFFFFu);                           // tile_dim1 (tile_dim2=0)
  g1[5] = (int)(unsigned)(s0 & 0xFFFFFFFFull);                 // tensor_dim0_stride lo
  g1[6] = (int)(unsigned)((s0 >> 32) & 0xFFFFull);             // stride hi (dim1 stride=0)
  g1[7] = 0;
  i32x4 z4 = {0, 0, 0, 0};
#if defined(__clang_major__) && (__clang_major__ >= 23)
  i32x8 z8 = {0, 0, 0, 0, 0, 0, 0, 0};
  __builtin_amdgcn_tensor_load_to_lds(g0, g1, z4, z4, z8, 0);
#else
  __builtin_amdgcn_tensor_load_to_lds(g0, g1, z4, z4, 0);
#endif
}

// ---- selection -------------------------------------------------------------
__global__ void keys_kernel(const float* __restrict__ c, unsigned* __restrict__ keys) {
  int i = blockIdx.x * blockDim.x + threadIdx.x;
  if (i < N_TOK) {
    unsigned u = __builtin_bit_cast(unsigned, c[i]);
    keys[i] = (u & 0x80000000u) ? ~u : (u | 0x80000000u);
  }
}

__global__ void select_top_kernel(const unsigned* __restrict__ keys,
                                  int* __restrict__ sel, int* __restrict__ mask) {
  int i = blockIdx.x * blockDim.x + threadIdx.x;
  if (i >= N_TOK) return;
  unsigned ki = keys[i];
  int rank = 0;
  for (int j = 0; j < N_TOK; ++j) {
    unsigned kj = keys[j];
    rank += (kj > ki) || (kj == ki && j < i);
  }
  if (rank < NTOP) { sel[rank] = i; mask[i] = 1; }
  else             { mask[i] = 0; }
}

__device__ __forceinline__ unsigned hash_u(unsigned x) {
  x ^= x >> 16; x *= 0x7feb352du; x ^= x >> 15; x *= 0x846ca68bu; x ^= x >> 16;
  return x;
}

__global__ void select_rand_kernel(const int* __restrict__ mask, int* __restrict__ sel) {
  int i = blockIdx.x * blockDim.x + threadIdx.x;
  if (i >= N_TOK) return;
  if (mask[i]) return;
  unsigned ri = hash_u((unsigned)i * 2654435761u + 12345u);
  int rank = 0;
  for (int j = 0; j < N_TOK; ++j) {
    if (mask[j]) continue;
    unsigned rj = hash_u((unsigned)j * 2654435761u + 12345u);
    rank += (rj > ri) || (rj == ri && j < i);
  }
  if (rank < NRAND) sel[NTOP + rank] = i;
}

// ---- layernorm -> bf16 (one 256-thread block per 512-wide row) -------------
__global__ void layernorm_bf16_kernel(const float* __restrict__ in,
                                      const float* __restrict__ g,
                                      const float* __restrict__ b,
                                      bf16* __restrict__ out, int ld) {
  __shared__ float sbuf[256];
  int row = blockIdx.x;
  const float* xr = in + (long)row * ld;
  int t = threadIdx.x;
  float v0 = xr[t], v1 = xr[t + 256];
  sbuf[t] = v0 + v1;
  __syncthreads();
  for (int s = 128; s > 0; s >>= 1) {
    if (t < s) sbuf[t] += sbuf[t + s];
    __syncthreads();
  }
  float mu = sbuf[0] * (1.0f / DMODEL);
  __syncthreads();
  float d0 = v0 - mu, d1 = v1 - mu;
  sbuf[t] = d0 * d0 + d1 * d1;
  __syncthreads();
  for (int s = 128; s > 0; s >>= 1) {
    if (t < s) sbuf[t] += sbuf[t + s];
    __syncthreads();
  }
  float inv = rsqrtf(sbuf[0] * (1.0f / DMODEL) + 1e-5f);
  out[(long)row * DMODEL + t]       = f2bf(d0 * inv * g[t] + b[t]);
  out[(long)row * DMODEL + t + 256] = f2bf(d1 * inv * g[t + 256] + b[t + 256]);
}

// ---- weight cast+transpose: WT[n*K + k] = bf16(W[k*Ncols + n]) -------------
__global__ void cast_transpose_kernel(const float* __restrict__ W, bf16* __restrict__ WT,
                                      int K, int Ncols) {
  long idx = (long)blockIdx.x * blockDim.x + threadIdx.x;
  long total = (long)K * Ncols;
  if (idx < total) {
    int n = (int)(idx / K);
    int k = (int)(idx % K);
    WT[idx] = f2bf(W[(long)k * Ncols + n]);
  }
}

// ---- gather selected raw x rows to bf16 ------------------------------------
__global__ void gather_cast_kernel(const float* __restrict__ x, const int* __restrict__ sel,
                                   bf16* __restrict__ out) {
  int row = blockIdx.x;           // 0..KTOT-1
  int src = sel[row];
  for (int c0 = threadIdx.x; c0 < DMODEL; c0 += blockDim.x)
    out[(long)row * DMODEL + c0] = f2bf(x[(long)src * DMODEL + c0]);
}

// ---- TDM-fed, LDS double-buffered bf16 GEMM --------------------------------
// Block tile: 128(M) x 64(N) x 64(K). 8 waves; wave w owns rows w*16..w*16+15
// and all 64 cols (4 accumulators). Wave 0 drives the Tensor Data Mover:
// stage s+1 issued before waiting TENSORcnt for stage s (double buffer).
#define BM 128
#define BN 64
#define BK 64

template<bool RELU, bool TRANSOUT>
__global__ void gemm_bf16_tdm(const bf16* __restrict__ A, long lda, long aHeadStride,
                              const bf16* __restrict__ Bt, long ldbt, long btHeadStride,
                              bf16* __restrict__ out, long ldo, int outColHeadStride,
                              const float* __restrict__ bias,
                              int M, int Ncols, int K) {
  __shared__ bf16 As[2][BM * BK];   // 2 x 16 KB
  __shared__ bf16 Bs[2][BN * BK];   // 2 x  8 KB

  int head = blockIdx.y;
  A  += (long)head * aHeadStride;
  Bt += (long)head * btHeadStride;
  int colBase = head * outColHeadStride;

  int tilesNB = Ncols / BN;
  int bm = blockIdx.x / tilesNB;
  int bn = blockIdx.x % tilesNB;
  int waveId = threadIdx.x >> 5;
  int lane   = threadIdx.x & 31;

  const bf16* aBase = A  + (long)bm * BM * lda;
  const bf16* bBase = Bt + (long)bn * BN * ldbt;
  int stages = K / BK;

  if (waveId == 0) {
    tdm_load_2d((unsigned)(uintptr_t)&As[0][0], aBase, BK, BM, (unsigned long long)lda);
    tdm_load_2d((unsigned)(uintptr_t)&Bs[0][0], bBase, BK, BN, (unsigned long long)ldbt);
  }

  f32x8 acc[4];
#pragma unroll
  for (int t = 0; t < 4; ++t) acc[t] = (f32x8){0.f, 0.f, 0.f, 0.f, 0.f, 0.f, 0.f, 0.f};

  for (int s = 0; s < stages; ++s) {
    int buf = s & 1;
    if (waveId == 0) {
      if (s + 1 < stages) {
        int nb = (s + 1) & 1;
        tdm_load_2d((unsigned)(uintptr_t)&As[nb][0], aBase + (long)(s + 1) * BK, BK, BM,
                    (unsigned long long)lda);
        tdm_load_2d((unsigned)(uintptr_t)&Bs[nb][0], bBase + (long)(s + 1) * BK, BK, BN,
                    (unsigned long long)ldbt);
        __builtin_amdgcn_s_wait_tensorcnt(2);  // stage s complete, s+1 in flight
      } else {
        __builtin_amdgcn_s_wait_tensorcnt(0);  // last stage complete
      }
    }
    __syncthreads();   // LDS buffer `buf` ready for all waves
#pragma unroll
    for (int kk = 0; kk < BK; kk += 32) {
      bf16x16 a = load_a_frag(&As[buf][0], BK, waveId * 16, kk, lane);
#pragma unroll
      for (int t = 0; t < 4; ++t) {
        bf16x16 b = load_b_frag(&Bs[buf][0], BK, t * 16, kk, lane);
        acc[t] = __builtin_amdgcn_wmma_f32_16x16x32_bf16(false, a, false, b, (short)0,
                                                         acc[t], false, false);
      }
    }
    __syncthreads();   // done reading `buf` before TDM overwrites it (stage s+2)
  }

  int m = lane & 15, half = (lane >> 4) & 1;
#pragma unroll
  for (int t = 0; t < 4; ++t) {
    int col = colBase + bn * BN + t * 16 + m;
    float bv = bias ? bias[col] : 0.0f;
#pragma unroll
    for (int r = 0; r < 8; ++r) {
      int row = bm * BM + waveId * 16 + r + half * 8;
      float v = acc[t][r] + bv;
      if (RELU) v = fmaxf(v, 0.0f);
      if (TRANSOUT) out[(long)col * ldo + row] = f2bf(v);  // ldo = M for transposed out
      else          out[(long)row * ldo + col] = f2bf(v);
    }
  }
}

// ---- fp32-out GEMM with bias + residual (optionally gathered) --------------
__global__ void gemm_f32_kernel(const bf16* __restrict__ A, long lda,
                                const bf16* __restrict__ Bt, long ldbt,
                                float* __restrict__ out, long ldo,
                                const float* __restrict__ bias,
                                const float* __restrict__ resid, long ldr,
                                const int* __restrict__ gather,
                                int M, int Ncols, int K) {
  int wavesPerBlock = blockDim.x >> 5;
  int waveId = threadIdx.x >> 5;
  int lane   = threadIdx.x & 31;
  int tilesN = Ncols >> 4;
  int tile = blockIdx.x * wavesPerBlock + waveId;
  int row0 = (tile / tilesN) << 4;
  int col0 = (tile % tilesN) << 4;

  f32x8 acc = {0.f, 0.f, 0.f, 0.f, 0.f, 0.f, 0.f, 0.f};
  for (int k0 = 0; k0 < K; k0 += 32) {
    bf16x16 a = load_a_frag(A, lda, row0, k0, lane);
    bf16x16 b = load_b_frag(Bt, ldbt, col0, k0, lane);
    acc = __builtin_amdgcn_wmma_f32_16x16x32_bf16(false, a, false, b, (short)0, acc,
                                                  false, false);
  }
  int m = lane & 15, half = (lane >> 4) & 1;
  int col = col0 + m;
  float bv = bias ? bias[col] : 0.0f;
#pragma unroll
  for (int r = 0; r < 8; ++r) {
    int row = row0 + r + half * 8;
    int rr = gather ? gather[row] : row;
    float res = resid ? resid[(long)rr * ldr + col] : 0.0f;
    out[(long)row * ldo + col] = acc[r] + bv + res;
  }
}

// ---- fused scores + softmax: one block = 16 queries x 512 keys, one head ---
__global__ void attn_softmax_kernel(const bf16* __restrict__ q, const bf16* __restrict__ k,
                                    float* __restrict__ pOut, bf16* __restrict__ pT) {
  __shared__ float sm[16 * KTOT];   // 32 KB
  __shared__ float red[256];
  int h = blockIdx.y;
  int row0 = blockIdx.x * 16;
  int waveId = threadIdx.x >> 5;
  int lane   = threadIdx.x & 31;

  // each of 8 waves covers 64 keys = 4 tiles; K-dim = 64 -> 2 WMMA steps
  for (int ct = 0; ct < 4; ++ct) {
    int col0 = waveId * 64 + ct * 16;
    f32x8 acc = {0.f, 0.f, 0.f, 0.f, 0.f, 0.f, 0.f, 0.f};
    for (int k0 = 0; k0 < DKH; k0 += 32) {
      bf16x16 a = load_a_frag(q, DMODEL, row0, h * DKH + k0, lane);
      bf16x16 b = load_b_frag(k, DMODEL, col0, h * DKH + k0, lane);
      acc = __builtin_amdgcn_wmma_f32_16x16x32_bf16(false, a, false, b, (short)0, acc,
                                                    false, false);
    }
    int m = lane & 15, half = (lane >> 4) & 1;
#pragma unroll
    for (int r = 0; r < 8; ++r)
      sm[(r + half * 8) * KTOT + col0 + m] = acc[r] * 0.125f;  // 1/sqrt(64)
  }
  __syncthreads();

  // softmax over 512 keys: 16 threads per row, 32 keys each
  int row = threadIdx.x >> 4;
  int seg = threadIdx.x & 15;
  float mx = -3.4e38f;
  for (int j = 0; j < 32; ++j) mx = fmaxf(mx, sm[row * KTOT + seg * 32 + j]);
  red[threadIdx.x] = mx;
  __syncthreads();
  if (seg == 0) {
    float m2 = red[row * 16];
    for (int t = 1; t < 16; ++t) m2 = fmaxf(m2, red[row * 16 + t]);
    red[row * 16] = m2;
  }
  __syncthreads();
  mx = red[row * 16];
  __syncthreads();
  float s = 0.0f;
  for (int j = 0; j < 32; ++j) {
    float e = __expf(sm[row * KTOT + seg * 32 + j] - mx);
    sm[row * KTOT + seg * 32 + j] = e;
    s += e;
  }
  red[threadIdx.x] = s;
  __syncthreads();
  if (seg == 0) {
    float t = 0.0f;
    for (int i = 0; i < 16; ++i) t += red[row * 16 + i];
    red[row * 16] = t;
  }
  __syncthreads();
  float inv = 1.0f / red[row * 16];

  int n = row0 + row;
  for (int j = 0; j < 32; ++j) {
    int key = seg * 32 + j;
    float pv = sm[row * KTOT + key] * inv;
    pOut[((long)h * N_TOK + n) * KTOT + key] = pv;           // p: (H, N, KTOT) fp32
    pT[((long)(h * KTOT + key)) * N_TOK + n] = f2bf(pv);     // pT: (H, KTOT, N) bf16
  }
}

// ---------------------------------------------------------------------------
extern "C" void kernel_launch(void* const* d_in, const int* in_sizes, int n_in,
                              void* d_out, int out_size, void* d_ws, size_t ws_size,
                              hipStream_t stream) {
  (void)in_sizes; (void)n_in; (void)out_size; (void)ws_size;
  const float* x     = (const float*)d_in[0];
  const float* c     = (const float*)d_in[1];
  const float* wq    = (const float*)d_in[2];
  const float* bq    = (const float*)d_in[3];
  const float* wk    = (const float*)d_in[4];
  const float* bk    = (const float*)d_in[5];
  const float* wv    = (const float*)d_in[6];
  const float* bv    = (const float*)d_in[7];
  const float* wo    = (const float*)d_in[8];
  const float* bo    = (const float*)d_in[9];
  const float* ln1_g = (const float*)d_in[10];
  const float* ln1_b = (const float*)d_in[11];
  const float* ln2_g = (const float*)d_in[12];
  const float* ln2_b = (const float*)d_in[13];
  const float* w1    = (const float*)d_in[14];
  const float* b1    = (const float*)d_in[15];
  const float* w2    = (const float*)d_in[16];
  const float* b2    = (const float*)d_in[17];

  float* outp = (float*)d_out;                       // (KTOT, DMODEL) fp32
  float* pOut = outp + (size_t)KTOT * DMODEL;        // (H, N, KTOT) fp32

  char* wsp = (char*)d_ws;
  auto carve = [&](size_t bytes) -> void* {
    void* p = (void*)wsp;
    wsp += (bytes + 255) & ~(size_t)255;
    return p;
  };
  bf16*  nx   = (bf16*)carve((size_t)N_TOK * DMODEL * 2);
  bf16*  qb   = (bf16*)carve((size_t)N_TOK * DMODEL * 2);
  bf16*  vT   = (bf16*)carve((size_t)N_TOK * DMODEL * 2);   // (DMODEL, N) = per-head dk-major
  bf16*  kb   = (bf16*)carve((size_t)KTOT * DMODEL * 2);
  bf16*  topk = (bf16*)carve((size_t)KTOT * DMODEL * 2);
  bf16*  ctxr = (bf16*)carve((size_t)KTOT * DMODEL * 2);
  bf16*  n2   = (bf16*)carve((size_t)KTOT * DMODEL * 2);
  bf16*  ffn  = (bf16*)carve((size_t)KTOT * DFF * 2);
  bf16*  wqT  = (bf16*)carve((size_t)DMODEL * DMODEL * 2);
  bf16*  wkT  = (bf16*)carve((size_t)DMODEL * DMODEL * 2);
  bf16*  wvT  = (bf16*)carve((size_t)DMODEL * DMODEL * 2);
  bf16*  woT  = (bf16*)carve((size_t)DMODEL * DMODEL * 2);
  bf16*  w1T  = (bf16*)carve((size_t)DMODEL * DFF * 2);
  bf16*  w2T  = (bf16*)carve((size_t)DFF * DMODEL * 2);
  float* h1   = (float*)carve((size_t)KTOT * DMODEL * 4);
  unsigned* keys = (unsigned*)carve((size_t)N_TOK * 4);
  int*   mask = (int*)carve((size_t)N_TOK * 4);
  int*   sel  = (int*)carve((size_t)KTOT * 4);
  bf16*  pT   = (bf16*)carve((size_t)NHEAD * KTOT * N_TOK * 2);

  // 1) token selection
  keys_kernel       <<<N_TOK / 256, 256, 0, stream>>>(c, keys);
  select_top_kernel <<<N_TOK / 256, 256, 0, stream>>>(keys, sel, mask);
  select_rand_kernel<<<N_TOK / 256, 256, 0, stream>>>(mask, sel);

  // 2) LN1 -> nx (bf16)
  layernorm_bf16_kernel<<<N_TOK, 256, 0, stream>>>(x, ln1_g, ln1_b, nx, DMODEL);

  // 3) weights -> bf16 transposed (N-major) for contiguous B fragments
  cast_transpose_kernel<<<(DMODEL * DMODEL) / 256, 256, 0, stream>>>(wq, wqT, DMODEL, DMODEL);
  cast_transpose_kernel<<<(DMODEL * DMODEL) / 256, 256, 0, stream>>>(wk, wkT, DMODEL, DMODEL);
  cast_transpose_kernel<<<(DMODEL * DMODEL) / 256, 256, 0, stream>>>(wv, wvT, DMODEL, DMODEL);
  cast_transpose_kernel<<<(DMODEL * DMODEL) / 256, 256, 0, stream>>>(wo, woT, DMODEL, DMODEL);
  cast_transpose_kernel<<<(DMODEL * DFF) / 256, 256, 0, stream>>>(w1, w1T, DMODEL, DFF);
  cast_transpose_kernel<<<(DFF * DMODEL) / 256, 256, 0, stream>>>(w2, w2T, DFF, DMODEL);

  // 4) gather raw selected rows (K path + residual uses raw x)
  gather_cast_kernel<<<KTOT, 256, 0, stream>>>(x, sel, topk);

  // 5) projections (TDM-fed tiled GEMMs)
  //    q = nx@wq+bq : grid (16384/128)*(512/64) = 1024
  gemm_bf16_tdm<false, false><<<dim3(1024, 1), 256, 0, stream>>>(
      nx, DMODEL, 0, wqT, DMODEL, 0, qb, DMODEL, 0, bq, N_TOK, DMODEL, DMODEL);
  //    k = topk@wk+bk : grid (512/128)*(512/64) = 32
  gemm_bf16_tdm<false, false><<<dim3(32, 1), 256, 0, stream>>>(
      topk, DMODEL, 0, wkT, DMODEL, 0, kb, DMODEL, 0, bk, KTOT, DMODEL, DMODEL);
  //    v = nx@wv+bv, stored transposed (dk-major per head)
  gemm_bf16_tdm<false, true><<<dim3(1024, 1), 256, 0, stream>>>(
      nx, DMODEL, 0, wvT, DMODEL, 0, vT, N_TOK, 0, bv, N_TOK, DMODEL, DMODEL);

  // 6) fused scores + softmax -> p (fp32, output) and pT (bf16, for p^T v)
  attn_softmax_kernel<<<dim3(N_TOK / 16, NHEAD), 256, 0, stream>>>(qb, kb, pOut, pT);

  // 7) ctx = p^T @ v per head: A = pT (KTOT x N), Bt = vT (DK x N), K = 16384
  //    grid per head: (512/128)*(64/64) = 4
  gemm_bf16_tdm<false, false><<<dim3(4, NHEAD), 256, 0, stream>>>(
      pT, N_TOK, (long)KTOT * N_TOK, vT, N_TOK, (long)DKH * N_TOK,
      ctxr, DMODEL, DKH, nullptr, KTOT, DKH, N_TOK);

  // 8) h1 = gather(x) + ctxr @ wo^T + bo
  gemm_f32_kernel<<<dim3(128, 1), 256, 0, stream>>>(
      ctxr, DMODEL, woT, DMODEL, h1, DMODEL, bo, x, DMODEL, sel, KTOT, DMODEL, DMODEL);

  // 9) LN2 -> n2 (bf16)
  layernorm_bf16_kernel<<<KTOT, 256, 0, stream>>>(h1, ln2_g, ln2_b, n2, DMODEL);

  // 10) ffn = relu(n2 @ w1^T + b1) : grid (512/128)*(2048/64) = 128
  gemm_bf16_tdm<true, false><<<dim3(128, 1), 256, 0, stream>>>(
      n2, DMODEL, 0, w1T, DMODEL, 0, ffn, DFF, 0, b1, KTOT, DFF, DMODEL);

  // 11) out = h1 + ffn @ w2^T + b2
  gemm_f32_kernel<<<dim3(128, 1), 256, 0, stream>>>(
      ffn, DFF, w2T, DFF, outp, DMODEL, b2, h1, DMODEL, nullptr, KTOT, DMODEL, DFF);
}
